// GroundTruthPermutedGru_71416716198144
// MI455X (gfx1250) — matrix operand order repaired and checked by reading, not compiled, not run on hardware
//
#include <hip/hip_runtime.h>
#include <hip/hip_bf16.h>

// ---------------------------------------------------------------------------
// GRU scan, MI455X (gfx1250): bf16 WMMA (f32 accum) for all matmuls.
//   Phase 0: pre-mask (tril) + pre-pack 6 weight matrices into bf16 B-fragment
//            panel layout (1KB per 32x16 panel, per-lane contiguous 32B).
//   Phase 1: big parallel GEMM: XR/XZ/XN[t,b,j] = (x @ Wi*_tril) + b_i* * x
//   Phase 2: ONE persistent kernel (32 blocks, co-resident) runs all 512
//            recurrent steps with a device-scope atomic grid barrier between
//            steps; h@Wh* uses triangular-skipped bf16 WMMA K-loops, fused
//            GRU elementwise + label threshold -> ans (any >= 0.5).
// ---------------------------------------------------------------------------

typedef __attribute__((ext_vector_type(16))) __bf16 v16bf;
typedef __attribute__((ext_vector_type(8)))  float  v8f;

#define TT   512
#define BB   64
#define HH   1024
#define KP   32          // K per bf16 WMMA
#define NPAN (HH / KP)   // 32 K-panels
#define JT   (HH / 16)   // 64 column tiles
#define NBLK 32          // persistent-scan blocks (8 waves each -> 256 waves)

__device__ __forceinline__ float sigm(float v) {
    return 1.0f / (1.0f + __expf(-v));
}

// Load a 16x32 fp32 A-tile (row-major, stride ld) into bf16 A-fragment layout.
// lane<16: rows M=lane, K = k0+{0..7,16..23}; lane>=16: rows M=lane-16,
// K = k0+{8..15,24..31}.
__device__ __forceinline__ v16bf load_a_frag(const float* __restrict__ src,
                                             int ld, int rowBase, int k0, int lane) {
    const int r   = lane & 15;
    const int off = (lane >= 16) ? 8 : 0;
    const float* p = src + (size_t)(rowBase + r) * ld + k0 + off;
    float4 f0 = *(const float4*)(p);
    float4 f1 = *(const float4*)(p + 4);
    float4 f2 = *(const float4*)(p + 16);
    float4 f3 = *(const float4*)(p + 20);
    v16bf a;
    a[0]  = (__bf16)f0.x;  a[1]  = (__bf16)f0.y;  a[2]  = (__bf16)f0.z;  a[3]  = (__bf16)f0.w;
    a[4]  = (__bf16)f1.x;  a[5]  = (__bf16)f1.y;  a[6]  = (__bf16)f1.z;  a[7]  = (__bf16)f1.w;
    a[8]  = (__bf16)f2.x;  a[9]  = (__bf16)f2.y;  a[10] = (__bf16)f2.z;  a[11] = (__bf16)f2.w;
    a[12] = (__bf16)f3.x;  a[13] = (__bf16)f3.y;  a[14] = (__bf16)f3.z;  a[15] = (__bf16)f3.w;
    return a;
}

__device__ __forceinline__ v8f wmma_bf16(v16bf a, v16bf b, v8f c) {
    return __builtin_amdgcn_wmma_f32_16x16x32_bf16(
        /*neg_a=*/false, a, /*neg_b=*/false, b,
        /*c_mod=*/(short)0, c, /*reuse_a=*/false, /*reuse_b=*/false);
}

// ---------------------------------------------------------------------------
// Phase 0a: ans init + barrier counter init.
__global__ void init_kernel(int* __restrict__ ans, int n, int* __restrict__ bar) {
    int i = blockIdx.x * blockDim.x + threadIdx.x;
    if (i < n) ans[i] = -1;
    if (i == 0) *bar = 0;
}

// Phase 0b: mask (tril) + pack one HxH fp32 matrix into bf16 B-panel layout.
// dst linear index == ((pk*JT + jt)*512 + lane*16 + e); element e of lane
// holds W[k][j], k = pk*32 + (e<8 ? 8*(lane>>4)+e : 16+8*(lane>>4)+(e-8)), j = jt*16 + (lane&15),
// zeroed when j > k (tril mask).
__global__ void pack_weight_kernel(const float* __restrict__ W,
                                   __bf16* __restrict__ dst) {
    int tid  = blockIdx.x * blockDim.x + threadIdx.x;  // 0 .. HH*HH-1
    int pk   = tid >> 15;          // / (JT*512)
    int rem  = tid & 32767;
    int jt   = rem >> 9;
    int rem2 = rem & 511;
    int lane = rem2 >> 4;
    int e    = rem2 & 15;
    int k = pk * KP + ((e < 8) ? (lane >> 4) * 8 + e : 16 + (lane >> 4) * 8 + (e - 8));
    int j = jt * 16 + (lane & 15);
    float v = (j <= k) ? W[k * HH + j] : 0.0f;
    dst[tid] = (__bf16)v;
}

// ---------------------------------------------------------------------------
// Phase 1: XR/XZ/XN = x @ (Wi*_tril) + b_i* * x, over all T*B rows.
// One wave per 16x16 output tile, computing all three matrices (A reuse).
__global__ void input_proj_kernel(const float* __restrict__ x,
                                  const __bf16* __restrict__ pWir,
                                  const __bf16* __restrict__ pWiz,
                                  const __bf16* __restrict__ pWin,
                                  const float* __restrict__ b_ir,
                                  const float* __restrict__ b_iz,
                                  const float* __restrict__ b_in,
                                  float* __restrict__ XR,
                                  float* __restrict__ XZ,
                                  float* __restrict__ XN) {
    const int lane = threadIdx.x & 31;
    const int wid  = blockIdx.x * (blockDim.x >> 5) + (threadIdx.x >> 5);
    const int jt   = wid & (JT - 1);
    const int rt   = wid >> 6;              // 0 .. T*B/16 - 1
    const int rowBase = rt * 16;
    const int j0      = jt * 16;

    v8f ar = {}, az = {}, an = {};
    const int p0 = jt >> 1;                 // triangular: only k >= j0 matters
#pragma unroll 2
    for (int p = p0; p < NPAN; ++p) {
        v16bf a  = load_a_frag(x, HH, rowBase, p * KP, lane);
        const size_t pb = ((size_t)(p * JT + jt)) * 512 + (size_t)lane * 16;
        v16bf br = *(const v16bf*)(pWir + pb);
        v16bf bz = *(const v16bf*)(pWiz + pb);
        v16bf bn = *(const v16bf*)(pWin + pb);
        ar = wmma_bf16(a, br, ar);
        az = wmma_bf16(a, bz, az);
        an = wmma_bf16(a, bn, an);
    }

    const int n  = lane & 15;
    const int mo = (lane >= 16) ? 8 : 0;
#pragma unroll
    for (int v = 0; v < 8; ++v) {
        int row = rowBase + v + mo;
        int j   = j0 + n;
        size_t o = (size_t)row * HH + j;
        float xv = x[o];
        XR[o] = ar[v] + b_ir[j] * xv;
        XZ[o] = az[v] + b_iz[j] * xv;
        XN[o] = an[v] + b_in[j] * xv;
    }
}

// ---------------------------------------------------------------------------
// Phase 2: persistent scan kernel. 256 waves (4 row tiles x 64 col tiles),
// each wave owns one 16x16 output tile for ALL 512 steps. Grid-wide sync
// between steps via monotonic agent-scope atomic counter; every wave performs
// the acquire spin so each CU's L0 is invalidated before re-reading h.
__global__ void gru_scan_kernel(const float* __restrict__ hidden0,
                                float* __restrict__ hb,
                                const float* __restrict__ x,
                                const __bf16* __restrict__ pWhr,
                                const __bf16* __restrict__ pWhz,
                                const __bf16* __restrict__ pWhn,
                                const float* __restrict__ XR,
                                const float* __restrict__ XZ,
                                const float* __restrict__ XN,
                                const float* __restrict__ b_hr,
                                const float* __restrict__ b_hz,
                                const float* __restrict__ b_hn,
                                const float* __restrict__ W_out,
                                const float* __restrict__ b_out,
                                int* __restrict__ ans,
                                int* __restrict__ bar) {
    const int lane = threadIdx.x & 31;
    const int wid  = blockIdx.x * (blockDim.x >> 5) + (threadIdx.x >> 5);
    const int jt   = wid & (JT - 1);
    const int rt   = wid >> 6;              // 0..3 (B/16)
    const int rowBase = rt * 16;
    const int p0      = jt >> 1;            // triangular K start

    // Per-lane column is fixed for the whole scan: hoist invariants.
    const int n  = lane & 15;
    const int mo = (lane >= 16) ? 8 : 0;
    const int j  = jt * 16 + n;
    const float bhr = b_hr[j];
    const float bhz = b_hz[j];
    const float bhn = b_hn[j];
    const float wo  = W_out[j];
    const float bo  = b_out[0];

    for (int t = 0; t < TT; ++t) {
        const float* h_in = (t == 0) ? hidden0
                                     : (hb + (size_t)((t - 1) & 1) * BB * HH);
        float* h_out = hb + (size_t)(t & 1) * BB * HH;

        v8f ar = {}, az = {}, an = {};
#pragma unroll 2
        for (int p = p0; p < NPAN; ++p) {
            v16bf a  = load_a_frag(h_in, HH, rowBase, p * KP, lane);
            const size_t pb = ((size_t)(p * JT + jt)) * 512 + (size_t)lane * 16;
            v16bf br = *(const v16bf*)(pWhr + pb);
            v16bf bz = *(const v16bf*)(pWhz + pb);
            v16bf bn = *(const v16bf*)(pWhn + pb);
            ar = wmma_bf16(a, br, ar);
            az = wmma_bf16(a, bz, az);
            an = wmma_bf16(a, bn, an);
        }

#pragma unroll
        for (int v = 0; v < 8; ++v) {
            int b = rowBase + v + mo;
            size_t hidx = (size_t)b * HH + j;
            size_t xidx = (size_t)t * (BB * HH) + hidx;
            float rr = sigm(XR[xidx] + ar[v] + bhr);
            float zz = sigm(XZ[xidx] + az[v] + bhz);
            float nn = tanhf(XN[xidx] + rr * (an[v] + bhn));
            float ho = h_in[hidx];
            float hn2 = ho * zz + (1.0f - zz) * nn;
            h_out[hidx] = hn2;
            float label = sigm(hn2 * wo + bo) * x[xidx];
            if (label >= 0.5f) ans[t * BB + b] = 1;  // any>=0.5 <=> max>=0.5
        }

        // ----- device-wide barrier between steps -----
        __threadfence();                 // make this thread's h_out visible
        __syncthreads();                 // all threads in block fenced
        if (threadIdx.x == 0) {
            __hip_atomic_fetch_add(bar, 1, __ATOMIC_RELEASE,
                                   __HIP_MEMORY_SCOPE_AGENT);
        }
        const int target = (int)gridDim.x * (t + 1);
        while (__hip_atomic_load(bar, __ATOMIC_ACQUIRE,
                                 __HIP_MEMORY_SCOPE_AGENT) < target) {
            __builtin_amdgcn_s_sleep(2);
        }
        __syncthreads();
    }
}

// ---------------------------------------------------------------------------
extern "C" void kernel_launch(void* const* d_in, const int* in_sizes, int n_in,
                              void* d_out, int out_size, void* d_ws, size_t ws_size,
                              hipStream_t stream) {
    (void)in_sizes; (void)n_in; (void)out_size; (void)ws_size;

    const float* x       = (const float*)d_in[0];   // [T,B,H]
    const float* hidden0 = (const float*)d_in[1];   // [B,H]
    const float* W_ir    = (const float*)d_in[2];
    const float* W_hr    = (const float*)d_in[3];
    const float* W_iz    = (const float*)d_in[4];
    const float* W_hz    = (const float*)d_in[5];
    const float* W_in    = (const float*)d_in[6];
    const float* W_hn    = (const float*)d_in[7];
    const float* b_ir    = (const float*)d_in[8];
    const float* b_hr    = (const float*)d_in[9];
    const float* b_iz    = (const float*)d_in[10];
    const float* b_hz    = (const float*)d_in[11];
    const float* b_in    = (const float*)d_in[12];
    const float* b_hn    = (const float*)d_in[13];
    const float* W_out   = (const float*)d_in[14];
    const float* b_out   = (const float*)d_in[15];
    int* ans = (int*)d_out;                         // [T,B] int32

    // Workspace layout
    const size_t MAT = (size_t)HH * HH;             // 1M elements
    char* wsb = (char*)d_ws;
    __bf16* pW  = (__bf16*)wsb;                     // 6 packed matrices, 12 MB
    __bf16* pWir = pW + 0 * MAT;
    __bf16* pWiz = pW + 1 * MAT;
    __bf16* pWin = pW + 2 * MAT;
    __bf16* pWhr = pW + 3 * MAT;
    __bf16* pWhz = pW + 4 * MAT;
    __bf16* pWhn = pW + 5 * MAT;
    float* XR = (float*)(wsb + 6 * MAT * sizeof(__bf16));      // 128 MB
    float* XZ = XR + (size_t)TT * BB * HH;                     // 128 MB
    float* XN = XZ + (size_t)TT * BB * HH;                     // 128 MB
    float* hb = XN + (size_t)TT * BB * HH;                     // 2 x 256 KB
    int* bar  = (int*)(hb + 2 * (size_t)BB * HH);              // barrier counter

    // Phase 0: ans/barrier init + weight pack (tril mask applied here)
    init_kernel<<<(TT * BB + 255) / 256, 256, 0, stream>>>(ans, TT * BB, bar);
    const int packBlocks = (int)(MAT / 256);
    pack_weight_kernel<<<packBlocks, 256, 0, stream>>>(W_ir, pWir);
    pack_weight_kernel<<<packBlocks, 256, 0, stream>>>(W_iz, pWiz);
    pack_weight_kernel<<<packBlocks, 256, 0, stream>>>(W_in, pWin);
    pack_weight_kernel<<<packBlocks, 256, 0, stream>>>(W_hr, pWhr);
    pack_weight_kernel<<<packBlocks, 256, 0, stream>>>(W_hz, pWhz);
    pack_weight_kernel<<<packBlocks, 256, 0, stream>>>(W_hn, pWhn);

    // Phase 1: parallel input projections. waves = (T*B/16) * (H/16) = 131072
    // -> 16384 blocks of 8 waves.
    input_proj_kernel<<<16384, 256, 0, stream>>>(x, pWir, pWiz, pWin,
                                                 b_ir, b_iz, b_in, XR, XZ, XN);

    // Phase 2: ONE persistent kernel runs the entire 512-step scan.
    // 32 blocks x 8 waves = 256 waves (4 row tiles x 64 col tiles),
    // trivially co-resident on MI455X.
    gru_scan_kernel<<<NBLK, 256, 0, stream>>>(hidden0, hb, x,
                                              pWhr, pWhz, pWhn,
                                              XR, XZ, XN,
                                              b_hr, b_hz, b_hn,
                                              W_out, b_out, ans, bar);
}
